// MultiHeadAttention_47682726920550
// MI455X (gfx1250) — matrix operand order repaired and checked by reading, not compiled, not run on hardware
//
#include <hip/hip_runtime.h>
#include <stdint.h>

// ---------------------------------------------------------------------------
// Types for CDNA5 WMMA (wave32): V_WMMA_F32_16X16X32_BF16
// ---------------------------------------------------------------------------
typedef __bf16 bh;
typedef __attribute__((ext_vector_type(16))) __bf16 v16bf;
typedef __attribute__((ext_vector_type(8)))  __bf16 v8bf;
typedef __attribute__((ext_vector_type(8)))  float  v8f;

union F8   { v8f  v; float f[8]; };
union F16B { v16bf v; v8bf h[2]; };

__device__ __forceinline__ v8f wmma_bf16(v16bf a, v16bf b, v8f c) {
    // D = A(16x32 bf16) * B(32x16 bf16) + C(16x16 f32)
    return __builtin_amdgcn_wmma_f32_16x16x32_bf16(
        /*neg_a=*/false, a, /*neg_b=*/false, b,
        /*c_mod=*/(short)0, c, /*reuse_a=*/false, /*reuse_b=*/false);
}

// A-fragment (16x32 bf16, row-major source, leading dim `ld`):
// lane L: row M = L%16; K = (L/16)*16 + {0..15}
__device__ __forceinline__ v16bf load_a_frag(const bh* base, int ld, int lane) {
    int m  = lane & 15;
    int kb = (lane >> 4) << 4;
    const bh* p = base + (size_t)m * ld + kb;
    F16B f;
    f.h[0] = *(const v8bf*)(p);        // K = kb .. kb+7
    f.h[1] = *(const v8bf*)(p + 8);    // K = kb+8 .. kb+15
    return f.v;
}

// generic->LDS: low 32 bits of the shared aperture address are the
// wave-relative LDS byte address (ISA 10.2 aperture mapping)
__device__ __forceinline__ uint32_t lds_off32(const void* p) {
    return (uint32_t)(uintptr_t)p;
}

// async copy 32B (one B fragment per lane) global -> LDS; tracked by ASYNCcnt.
// INST_OFFSET is added to BOTH the LDS and the global address (ISA 10.2/15.18).
__device__ __forceinline__ void async_copy_b32B(uint32_t lds_dst, const void* gsrc) {
    uint64_t g = (uint64_t)(uintptr_t)gsrc;
    asm volatile("global_load_async_to_lds_b128 %0, %1, off"
                 :: "v"(lds_dst), "v"(g) : "memory");
    asm volatile("global_load_async_to_lds_b128 %0, %1, off offset:16"
                 :: "v"(lds_dst), "v"(g) : "memory");
}

// ---------------------------------------------------------------------------
// Kernel 1: fp32 -> bf16 convert
// ---------------------------------------------------------------------------
__global__ void cvt_bf16_kernel(const float* __restrict__ in, bh* __restrict__ out, int n) {
    int i = blockIdx.x * blockDim.x + threadIdx.x;
    if (i < n) out[i] = (bh)in[i];
}

// ---------------------------------------------------------------------------
// Kernel 2: repack weight W[K=768][N=768] (row-major fp32) into WMMA B-fragment
// order: tile = nt*ktiles + kt ; dst[((tile)*32 + lane)*16 + e]
// lane L: col N = nt*16 + L%16 ; element e: K = kt*32 + (L/16)*16 + e
// ---------------------------------------------------------------------------
__global__ void pack_w_kernel(const float* __restrict__ W, bh* __restrict__ Wp,
                              int K, int N) {
    int tid  = blockIdx.x * blockDim.x + threadIdx.x;
    int lane = tid & 31;
    int tile = tid >> 5;
    int ktiles = K >> 5;
    int nt = tile / ktiles;
    int kt = tile % ktiles;
    if (nt * 16 >= N) return;
    int n  = nt * 16 + (lane & 15);
    int k0 = kt * 32 + ((lane >> 4) << 4);
    bh* dst = Wp + ((size_t)tile * 32 + lane) * 16;
#pragma unroll
    for (int e = 0; e < 16; ++e)
        dst[e] = (bh)W[(size_t)(k0 + e) * N + n];
}

// ---------------------------------------------------------------------------
// Kernel 3: bf16 WMMA GEMM  C[M,N] = A[M,K] * B[K,N]
// - 4 waves/block; all 4 waves share nt (same B tile).
// - B tile (4KB) double-buffered in LDS via GLOBAL_LOAD_ASYNC_TO_LDS_B128
//   (each wave async-stages its quarter of the NEXT k-tile while the current
//   k-tile's 8 WMMAs run); sync = s_wait_asynccnt 0 + workgroup barrier.
// - A fragments register double-buffered.
// One wave computes a 32x64 tile (2x4 f32 accumulators).
// ---------------------------------------------------------------------------
__global__ __launch_bounds__(128)
void gemm_wmma_kernel(const bh* __restrict__ A, const bh* __restrict__ Bp,
                      float* __restrict__ Cout, int M, int N, int K) {
    __shared__ __align__(32) bh bbuf[2][4 * 32 * 16];   // 2 x 4KB B tiles

    int lane = threadIdx.x & 31;
    int w    = threadIdx.x >> 5;
    int wid  = blockIdx.x * 4 + w;
    int mTiles = M >> 5;                    // 32-row wave tiles
    int mt = wid % mTiles;
    int nt = wid / mTiles;                  // 64-col tile, shared by the block
    const int ktiles = K >> 5;

    const bh* aBase0 = A + (size_t)mt * 32 * K;
    const bh* aBase1 = aBase0 + (size_t)16 * K;
    // wave w stages fragment-column j=w of the shared B tile
    const bh* bSrc   = Bp + (((size_t)(nt * 4 + w) * ktiles) * 32 + lane) * 16;
    const uint32_t ldsW      = lds_off32(&bbuf[0][(w * 32 + lane) * 16]);
    const uint32_t ldsStride = (uint32_t)(4 * 32 * 16 * sizeof(bh));   // 4KB

    // prologue: stage k-tile 0 into buffer 0, A fragments for kt=0
    async_copy_b32B(ldsW, bSrc);
    v16bf a0 = load_a_frag(aBase0, K, lane);
    v16bf a1 = load_a_frag(aBase1, K, lane);

    F8 acc[2][4];
#pragma unroll
    for (int i = 0; i < 2; ++i)
#pragma unroll
        for (int j = 0; j < 4; ++j) acc[i][j].v = (v8f)0.0f;

    for (int kt = 0; kt < ktiles; ++kt) {
        // own async stage of buf[kt&1] done, then barrier -> whole tile visible
        asm volatile("s_wait_asynccnt 0x0" ::: "memory");
        __syncthreads();

        v16bf na0 = a0, na1 = a1;
        if (kt + 1 < ktiles) {
            // stage NEXT B tile (other buffer) + next A fragments: overlaps WMMA
            async_copy_b32B(ldsW + (uint32_t)((kt + 1) & 1) * ldsStride,
                            bSrc + (size_t)(kt + 1) * 32 * 16);
            na0 = load_a_frag(aBase0 + (kt + 1) * 32, K, lane);
            na1 = load_a_frag(aBase1 + (kt + 1) * 32, K, lane);
            __builtin_prefetch(aBase0 + (size_t)(lane & 15) * K + (kt + 1) * 32, 0, 3);
        }

        const bh* bl = &bbuf[kt & 1][0];
#pragma unroll
        for (int j = 0; j < 4; ++j) {
            v16bf bfr = *(const v16bf*)(bl + (j * 32 + lane) * 16);   // ds_load_b128 x2
            acc[0][j].v = wmma_bf16(a0, bfr, acc[0][j].v);
            acc[1][j].v = wmma_bf16(a1, bfr, acc[1][j].v);
        }
        a0 = na0; a1 = na1;
    }

    int g = lane >> 4, n0 = lane & 15;
#pragma unroll
    for (int i = 0; i < 2; ++i)
#pragma unroll
        for (int r = 0; r < 8; ++r) {
            int row = mt * 32 + i * 16 + g * 8 + r;
#pragma unroll
            for (int j = 0; j < 4; ++j)
                Cout[(size_t)row * N + nt * 64 + j * 16 + n0] = acc[i][j].f[r];
        }
}

// ---------------------------------------------------------------------------
// Kernel 4: RoPE + RMS-norm on q,k ; layout shuffle for v.
// One wave per (b, t, h). Lane i handles rotation pair (d=i, d=i+32), d<32.
// Outputs: qb/kb head-major [b][h][T][64] bf16 ; vt dim-major [b][h][64][T] bf16.
// ---------------------------------------------------------------------------
__global__ __launch_bounds__(256)
void rope_rms_kernel(const float* __restrict__ qf, const float* __restrict__ kf,
                     const float* __restrict__ vf,
                     const float* __restrict__ cosp, const float* __restrict__ sinp,
                     bh* __restrict__ qb, bh* __restrict__ kb, bh* __restrict__ vt) {
    const int T = 2048, H = 12, C = 768, D = 64;
    int wid  = blockIdx.x * (blockDim.x >> 5) + (threadIdx.x >> 5);
    int lane = threadIdx.x & 31;
    int h = wid % H;
    int t = (wid / H) % T;
    int b = wid / (H * T);

    size_t row = (size_t)(b * T + t) * C + h * D;
    float c = cosp[t * 32 + lane];
    float s = sinp[t * 32 + lane];
    size_t ob = ((size_t)(b * H + h) * T + t) * D;

    {   // q: rotary + rms
        float x1 = qf[row + lane], x2 = qf[row + 32 + lane];
        float y1 =  x1 * c + x2 * s;
        float y2 = -x1 * s + x2 * c;
        float ss = y1 * y1 + y2 * y2;
        for (int m = 1; m < 32; m <<= 1) ss += __shfl_xor(ss, m);
        float inv = rsqrtf(ss * (1.0f / 64.0f) + 1e-6f);
        qb[ob + lane]      = (bh)(y1 * inv);
        qb[ob + 32 + lane] = (bh)(y2 * inv);
    }
    {   // k: rotary + rms
        float x1 = kf[row + lane], x2 = kf[row + 32 + lane];
        float y1 =  x1 * c + x2 * s;
        float y2 = -x1 * s + x2 * c;
        float ss = y1 * y1 + y2 * y2;
        for (int m = 1; m < 32; m <<= 1) ss += __shfl_xor(ss, m);
        float inv = rsqrtf(ss * (1.0f / 64.0f) + 1e-6f);
        kb[ob + lane]      = (bh)(y1 * inv);
        kb[ob + 32 + lane] = (bh)(y2 * inv);
    }
    {   // v: transpose into [b][h][64][T]
        float v1 = vf[row + lane], v2 = vf[row + 32 + lane];
        size_t vb = (size_t)(b * H + h) * D;
        vt[(vb + lane) * T + t]      = (bh)v1;
        vt[(vb + 32 + lane) * T + t] = (bh)v2;
    }
}

// ---------------------------------------------------------------------------
// Kernel 5: flash attention. One wave per (b, h, 16-query tile).
// Per 32-key block: K AND V fragments loaded up front (V loads hide behind the
// softmax VALU work), 4 WMMA (S), online softmax (16-lane shfl reductions),
// P transposed C-layout -> A-layout through LDS, 4 WMMA (O += P*V).
// ---------------------------------------------------------------------------
__global__ __launch_bounds__(128)
void attn_fa_kernel(const bh* __restrict__ qb, const bh* __restrict__ kb,
                    const bh* __restrict__ vt, bh* __restrict__ y0) {
    const int T = 2048, H = 12, D = 64, C = 768;
    __shared__ __align__(32) bh pl[4][16 * 32];

    int w    = threadIdx.x >> 5;
    int lane = threadIdx.x & 31;
    int wid  = blockIdx.x * 4 + w;
    int qt = wid & 127;                 // T/16 = 128 query tiles
    int h  = (wid >> 7) % H;
    int b  = wid / (128 * H);
    int n0 = lane & 15;
    int g  = lane >> 4;

    const bh* qh = qb + ((size_t)(b * H + h) * T + qt * 16) * D;
    const bh* kh = kb + ((size_t)(b * H + h) * T) * D;
    const bh* vh = vt + ((size_t)(b * H + h) * D) * T;

    v16bf qA0 = load_a_frag(qh, D, lane);        // dims 0..31
    v16bf qA1 = load_a_frag(qh + 32, D, lane);   // dims 32..63

    F8 acc[4];
#pragma unroll
    for (int j = 0; j < 4; ++j) acc[j].v = (v8f)0.0f;
    float mrow[8], lrow[8];
#pragma unroll
    for (int r = 0; r < 8; ++r) { mrow[r] = -1e30f; lrow[r] = 0.0f; }

    const float scale = 0.125f;   // 1/sqrt(64)

    for (int kv0 = 0; kv0 < T; kv0 += 32) {
        // --- issue ALL global loads for this block up front ---------------
        v16bf k00 = *(const v16bf*)(kh + (size_t)(kv0 +      n0) * D +      g * 16);
        v16bf k10 = *(const v16bf*)(kh + (size_t)(kv0 +      n0) * D + 32 + g * 16);
        v16bf k01 = *(const v16bf*)(kh + (size_t)(kv0 + 16 + n0) * D +      g * 16);
        v16bf k11 = *(const v16bf*)(kh + (size_t)(kv0 + 16 + n0) * D + 32 + g * 16);
        v16bf vB[4];
#pragma unroll
        for (int j = 0; j < 4; ++j)   // V frags: consumed only after softmax
            vB[j] = *(const v16bf*)(vh + (size_t)(j * 16 + n0) * T + kv0 + g * 16);
        if (kv0 + 32 < T) {           // WGP-scope prefetch of the next K block
            __builtin_prefetch(kh + (size_t)(kv0 + 32 + n0) * D, 0, 3);
        }

        F8 s0, s1;
        s0.v = wmma_bf16(qA0, k00, (v8f)0.0f);
        s0.v = wmma_bf16(qA1, k10, s0.v);
        s1.v = wmma_bf16(qA0, k01, (v8f)0.0f);
        s1.v = wmma_bf16(qA1, k11, s1.v);

        // Online softmax per row (rows M = g*8 + r live in the same 16-lane group)
#pragma unroll
        for (int r = 0; r < 8; ++r) {
            float a0 = s0.f[r] * scale, a1 = s1.f[r] * scale;
            float rm = fmaxf(a0, a1);
            for (int ml = 1; ml < 16; ml <<= 1)
                rm = fmaxf(rm, __shfl_xor(rm, ml));
            float mn = fmaxf(mrow[r], rm);
            float al = __expf(mrow[r] - mn);
            mrow[r] = mn;
            float p0 = __expf(a0 - mn);
            float p1 = __expf(a1 - mn);
            float rs = p0 + p1;
            for (int ml = 1; ml < 16; ml <<= 1)
                rs += __shfl_xor(rs, ml);
            lrow[r] = lrow[r] * al + rs;
#pragma unroll
            for (int j = 0; j < 4; ++j) acc[j].f[r] *= al;
            int m = g * 8 + r;
            pl[w][m * 32 +      n0] = (bh)p0;   // C-layout -> matrix position
            pl[w][m * 32 + 16 + n0] = (bh)p1;
        }
        asm volatile("s_wait_dscnt 0" ::: "memory");   // P stores visible to A-frag loads
        v16bf pA = load_a_frag(&pl[w][0], 32, lane);   // P as 16x32 A fragment
#pragma unroll
        for (int j = 0; j < 4; ++j)
            acc[j].v = wmma_bf16(pA, vB[j], acc[j].v);
    }

    // y0 bf16 row-major [B*T, C] (A-matrix input of the projection GEMM)
#pragma unroll
    for (int r = 0; r < 8; ++r) {
        float inv = 1.0f / lrow[r];
        int tok = qt * 16 + g * 8 + r;
        size_t rowb = ((size_t)b * T + tok) * C + h * D;
#pragma unroll
        for (int j = 0; j < 4; ++j)
            y0[rowb + j * 16 + n0] = (bh)(acc[j].f[r] * inv);
    }
}

// ---------------------------------------------------------------------------
// Host launch
// ---------------------------------------------------------------------------
extern "C" void kernel_launch(void* const* d_in, const int* in_sizes, int n_in,
                              void* d_out, int out_size, void* d_ws, size_t ws_size,
                              hipStream_t stream) {
    (void)in_sizes; (void)n_in; (void)out_size; (void)ws_size;
    const float* x    = (const float*)d_in[0];
    const float* cosp = (const float*)d_in[1];
    const float* sinp = (const float*)d_in[2];
    const float* wq   = (const float*)d_in[3];
    const float* wk   = (const float*)d_in[4];
    const float* wv   = (const float*)d_in[5];
    const float* wp   = (const float*)d_in[6];
    float* out = (float*)d_out;

    const int Bb = 2, T = 2048, C = 768;
    const size_t NT = (size_t)Bb * T;       // 4096 tokens

    char* ws = (char*)d_ws;
    bh* xb  = (bh*)ws;  ws += NT * C * sizeof(bh);
    bh* wqp = (bh*)ws;  ws += (size_t)C * C * sizeof(bh);
    bh* wkp = (bh*)ws;  ws += (size_t)C * C * sizeof(bh);
    bh* wvp = (bh*)ws;  ws += (size_t)C * C * sizeof(bh);
    bh* wpp = (bh*)ws;  ws += (size_t)C * C * sizeof(bh);
    float* qf = (float*)ws; ws += NT * C * sizeof(float);
    float* kf = (float*)ws; ws += NT * C * sizeof(float);
    float* vf = (float*)ws; ws += NT * C * sizeof(float);
    bh* qbf = (bh*)ws;  ws += NT * C * sizeof(bh);
    bh* kbf = (bh*)ws;  ws += NT * C * sizeof(bh);
    bh* vtb = (bh*)ws;  ws += NT * C * sizeof(bh);
    bh* y0b = (bh*)ws;  ws += NT * C * sizeof(bh);

    // 1) x -> bf16
    {
        int n = (int)(NT * C);
        cvt_bf16_kernel<<<(n + 255) / 256, 256, 0, stream>>>(x, xb, n);
    }
    // 2) pack weights into WMMA B-fragment layout
    {
        int threads = (C / 16) * (C / 32) * 32;          // 36864
        pack_w_kernel<<<threads / 256, 256, 0, stream>>>(wq, wqp, C, C);
        pack_w_kernel<<<threads / 256, 256, 0, stream>>>(wk, wkp, C, C);
        pack_w_kernel<<<threads / 256, 256, 0, stream>>>(wv, wvp, C, C);
        pack_w_kernel<<<threads / 256, 256, 0, stream>>>(wp, wpp, C, C);
    }
    // 3) QKV projections (fp32 outputs)
    {
        int waves  = (int)(NT / 32) * (C / 64);          // 1536
        int blocks = waves / 4;                          // 4 waves / block
        gemm_wmma_kernel<<<blocks, 128, 0, stream>>>(xb, wqp, qf, (int)NT, C, C);
        gemm_wmma_kernel<<<blocks, 128, 0, stream>>>(xb, wkp, kf, (int)NT, C, C);
        gemm_wmma_kernel<<<blocks, 128, 0, stream>>>(xb, wvp, vf, (int)NT, C, C);
    }
    // 4) RoPE + RMS-norm, v transpose (one wave per (b,t,h))
    {
        int waves  = (int)NT * 12;                       // 49152
        int blocks = waves / 8;                          // 8 waves / block of 256
        rope_rms_kernel<<<blocks, 256, 0, stream>>>(qf, kf, vf, cosp, sinp,
                                                    qbf, kbf, vtb);
    }
    // 5) flash attention (one wave per (b,h,qtile))
    {
        int waves  = Bb * 12 * (T / 16);                 // 3072
        attn_fa_kernel<<<waves / 4, 128, 0, stream>>>(qbf, kbf, vtb, y0b);
    }
    // 6) output projection -> d_out (fp32)
    {
        int waves  = (int)(NT / 32) * (C / 64);
        gemm_wmma_kernel<<<waves / 4, 128, 0, stream>>>(y0b, wpp, out, (int)NT, C, C);
    }
}
